// DebertaAttention_38448547234717
// MI455X (gfx1250) — compile-verified, matrix operand order checked by reading, not run on hardware
//
#include <hip/hip_runtime.h>

// ---------------------------------------------------------------------------
// DeBERTa disentangled attention block for MI455X (gfx1250, wave32, WMMA).
// - All GEMMs: v_wmma_f32_16x16x32_bf16, fp32 accumulate.
// - Projection GEMMs: double-buffered LDS tiles filled by the Tensor Data
//   Mover (tensor_load_to_lds + s_wait_tensorcnt), compute overlapped with
//   the DMA of the next K tile; waves read ds_load_b128 fragments from LDS.
// - Attention: fused flash-style per (b,h,16 q rows): ctx + banded c2p/p2c
//   WMMA mini-GEMMs -> LDS diagonal gather -> online softmax -> WMMA PV.
//   Never materializes (B,H,L,L) tensors in HBM (saves ~1.3 GB of traffic).
// ---------------------------------------------------------------------------

typedef __bf16 bf16;
typedef __attribute__((ext_vector_type(16))) __bf16 bf16x16;
typedef __attribute__((ext_vector_type(8)))  __bf16 bf16x8;
typedef __attribute__((ext_vector_type(8)))  float  f32x8;
typedef __attribute__((ext_vector_type(4)))  unsigned int u32x4;
typedef __attribute__((ext_vector_type(8)))  int i32x8;
typedef __attribute__((ext_vector_type(4)))  int i32x4;

#define NB 4
#define NL 1024
#define ND 1024
#define NH 16
#define NDH 64
#define NP 1024   /* 2*MAXLEN */
#define HD 1024   /* NH*NDH  */

#if __has_builtin(__builtin_amdgcn_tensor_load_to_lds) && \
    __has_builtin(__builtin_amdgcn_s_wait_tensorcnt)
#define HAS_TDM 1
#else
#define HAS_TDM 0
#endif

// Fragment loader matching the CDNA5 16-bit A/B WMMA VGPR layout:
// lanes 0-15 take K chunks {0..7, 16..23}, lanes 16-31 take {8..15, 24..31}.
// Caller passes p already offset by the lane's khalf; we read +0 and +16.
__device__ __forceinline__ bf16x16 load_frag(const bf16* p) {
  bf16x8 lo = *(const bf16x8*)(p);
  bf16x8 hi = *(const bf16x8*)(p + 16);
  bf16x16 r;
#pragma unroll
  for (int i = 0; i < 8; ++i) { r[i] = lo[i]; r[i + 8] = hi[i]; }
  return r;
}

__device__ __forceinline__ f32x8 wmma_bf16(bf16x16 a, bf16x16 b, f32x8 c) {
  return __builtin_amdgcn_wmma_f32_16x16x32_bf16(false, a, false, b,
                                                 (short)0, c, false, false);
}

#if HAS_TDM
// Build a 2D Tensor-DMA descriptor (D#) and issue TENSOR_LOAD_TO_LDS.
// group0: [1:0] count=1 | [63:32] lds_addr | [120:64] global_addr | [127:126] type=2
// group1: [17:16] data_size=1 (2B) | [79:48] tensor_dim0 | [111:80] tensor_dim1
//         | [127:112] tile_dim0 | [143:128] tile_dim1 | [207:160] dim0 stride
// All inputs are block-uniform so the descriptor lands in SGPRs.
// This toolchain exposes the 6-arg builtin:
//   (u32x4 g0, i32x8 g1, i32x4 g2, i32x4 g3, i32x8 g4, i32 cpol)
__device__ __forceinline__ void tdm_load_2d(const bf16* gsrc, bf16* ldst,
                                            int tile_rows, int tile_k,
                                            int tensor_k, int tensor_rows,
                                            int stride_elems) {
  unsigned long long ga = (unsigned long long)(__SIZE_TYPE__)gsrc;
  unsigned int la = (unsigned int)(__SIZE_TYPE__)ldst;  // flat LDS addr[31:0] = LDS offset
  u32x4 g0;
  g0[0] = 1u;                                           // count=1, user mode
  g0[1] = la;
  g0[2] = (unsigned int)(ga & 0xffffffffu);
  g0[3] = (unsigned int)((ga >> 32) & 0x01ffffffu) | 0x80000000u;  // type=2
  i32x8 g1;
  g1[0] = 1 << 16;                                      // data_size = 2 bytes
  g1[1] = (tensor_k & 0xffff) << 16;                    // tensor_dim0[15:0]
  g1[2] = ((unsigned)tensor_k >> 16) | ((tensor_rows & 0xffff) << 16);
  g1[3] = ((unsigned)tensor_rows >> 16) | ((tile_k & 0xffff) << 16);  // tile_dim0
  g1[4] = tile_rows & 0xffff;                           // tile_dim1 (tile_dim2=0)
  g1[5] = stride_elems;                                 // tensor_dim0_stride[31:0]
  g1[6] = 0;
  g1[7] = 0;
  i32x4 z4 = {0, 0, 0, 0};                              // groups 2/3 unused (2D)
  i32x8 z8 = {0, 0, 0, 0, 0, 0, 0, 0};
  __builtin_amdgcn_tensor_load_to_lds(g0, g1, z4, z4, z8, 0);
}
#endif

// ---------------------------------------------------------------------------
// fp32 -> bf16 convert (grid-stride)
// ---------------------------------------------------------------------------
__global__ __launch_bounds__(256) void cvt_f32_bf16(const float* __restrict__ s,
                                                    bf16* __restrict__ d, int n) {
  for (int i = blockIdx.x * blockDim.x + threadIdx.x; i < n;
       i += gridDim.x * blockDim.x)
    d[i] = (bf16)s[i];
}

// ---------------------------------------------------------------------------
// 1024x1024 fp32 -> bf16 transpose (LDS tiled). Makes every GEMM NT so
// B-fragment/TDM loads are contiguous.
// ---------------------------------------------------------------------------
__global__ __launch_bounds__(256) void transpose_cvt(const float* __restrict__ W,
                                                     bf16* __restrict__ Wt) {
  __shared__ float tile[32][33];
  int bx = blockIdx.x * 32, by = blockIdx.y * 32;
#pragma unroll
  for (int i = threadIdx.y; i < 32; i += 8)
    tile[i][threadIdx.x] = W[(size_t)(by + i) * ND + bx + threadIdx.x];
  __syncthreads();
#pragma unroll
  for (int i = threadIdx.y; i < 32; i += 8)
    Wt[(size_t)(bx + i) * ND + by + threadIdx.x] = (bf16)tile[threadIdx.x][i];
}

// ---------------------------------------------------------------------------
// NT GEMM: C[m,n] = sum_k A[m,k]*Bt[n,k].  A (MxK), Bt (NxK) bf16 row-major.
// Block = 8 waves = 128x64 C tile. K loop in 64-wide steps; the A(128x64) and
// B(64x64) tiles are DMA'd into double-buffered LDS by the TDM (issued by
// wave 0, tracked with TENSORcnt) while the waves compute the current tile.
// ---------------------------------------------------------------------------
template <bool F32OUT>
__global__ __launch_bounds__(256) void gemm_nt_wmma(const bf16* __restrict__ A,
                                                    const bf16* __restrict__ Bt,
                                                    void* __restrict__ Cout,
                                                    int M, int N, int K) {
  __shared__ __attribute__((aligned(16))) bf16 sA[2][128 * 64];
  __shared__ __attribute__((aligned(16))) bf16 sB[2][64 * 64];

  int wave = threadIdx.x >> 5;
  int lane = threadIdx.x & 31;
  int m0 = blockIdx.x * 128;
  int n0 = blockIdx.y * 64;
  int lr = lane & 15;
  int khalf = (lane >> 4) * 8;

  auto issue = [&](int buf, int k0) {
#if HAS_TDM
    if (threadIdx.x < 32) {
      tdm_load_2d(A  + (size_t)m0 * K + k0, &sA[buf][0], 128, 64, K, M - m0, K);
      tdm_load_2d(Bt + (size_t)n0 * K + k0, &sB[buf][0],  64, 64, K, N - n0, K);
    }
#else
    for (int t = threadIdx.x; t < 128 * 8; t += 256) {   // 16B chunks
      int r = t >> 3, c = (t & 7) * 8;
      *(bf16x8*)&sA[buf][r * 64 + c] =
          *(const bf16x8*)(A + (size_t)(m0 + r) * K + k0 + c);
    }
    for (int t = threadIdx.x; t < 64 * 8; t += 256) {
      int r = t >> 3, c = (t & 7) * 8;
      *(bf16x8*)&sB[buf][r * 64 + c] =
          *(const bf16x8*)(Bt + (size_t)(n0 + r) * K + k0 + c);
    }
#endif
  };

  f32x8 acc0 = {}, acc1 = {}, acc2 = {}, acc3 = {};
  int ksteps = K / 64;
  issue(0, 0);
  for (int ks = 0; ks < ksteps; ++ks) {
    int buf = ks & 1;
#if HAS_TDM
    if (threadIdx.x < 32) __builtin_amdgcn_s_wait_tensorcnt(0);
#endif
    __syncthreads();                       // tile ks visible to all waves
    if (ks + 1 < ksteps) issue(buf ^ 1, (ks + 1) * 64);  // overlap DMA w/ math
    const bf16* la = &sA[buf][0] + (size_t)(wave * 16 + lr) * 64;
    const bf16* lb = &sB[buf][0];
#pragma unroll
    for (int kk = 0; kk < 64; kk += 32) {
      bf16x16 a  = load_frag(la + kk + khalf);
      bf16x16 b0 = load_frag(lb + (size_t)( 0 + lr) * 64 + kk + khalf);
      bf16x16 b1 = load_frag(lb + (size_t)(16 + lr) * 64 + kk + khalf);
      bf16x16 b2 = load_frag(lb + (size_t)(32 + lr) * 64 + kk + khalf);
      bf16x16 b3 = load_frag(lb + (size_t)(48 + lr) * 64 + kk + khalf);
      acc0 = wmma_bf16(a, b0, acc0);
      acc1 = wmma_bf16(a, b1, acc1);
      acc2 = wmma_bf16(a, b2, acc2);
      acc3 = wmma_bf16(a, b3, acc3);
    }
  }
  // C fragment layout: VGPR r -> M = m0+wave*16 + r + (lane>=16?8:0), N = n0+nt*16+(lane&15)
  int rbase = m0 + wave * 16 + ((lane & 16) ? 8 : 0);
  f32x8 accs[4] = {acc0, acc1, acc2, acc3};
#pragma unroll
  for (int nt = 0; nt < 4; ++nt)
#pragma unroll
    for (int r = 0; r < 8; ++r) {
      size_t idx = (size_t)(rbase + r) * N + n0 + nt * 16 + lr;
      if (F32OUT) ((float*)Cout)[idx] = accs[nt][r];
      else        ((bf16*)Cout)[idx] = (bf16)accs[nt][r];
    }
}

// ---------------------------------------------------------------------------
// Head-split reshapes (bf16 -> bf16, pure layout):
//   split_qk: (B*L, H*DH) -> (B,H,L,DH)
//   split_v : (B*L, H*DH) -> (B,H,DH,L)   (V stored transposed for PV B-frags)
//   split_p : (P,   H*DH) -> (H,P,DH)
// ---------------------------------------------------------------------------
__global__ __launch_bounds__(256) void split_qk(const bf16* __restrict__ s,
                                                bf16* __restrict__ d) {
  size_t i = (size_t)blockIdx.x * blockDim.x + threadIdx.x;
  int dd = i & 63; size_t t = i >> 6; int h = t & 15; t >>= 4;
  int l = t & 1023; int b = (int)(t >> 10);
  d[(((size_t)(b * NH + h)) * NL + l) * NDH + dd] = s[i];
}
__global__ __launch_bounds__(256) void split_v(const bf16* __restrict__ s,
                                               bf16* __restrict__ d) {
  size_t i = (size_t)blockIdx.x * blockDim.x + threadIdx.x;
  int dd = i & 63; size_t t = i >> 6; int h = t & 15; t >>= 4;
  int l = t & 1023; int b = (int)(t >> 10);
  d[(((size_t)(b * NH + h)) * NDH + dd) * NL + l] = s[i];
}
__global__ __launch_bounds__(256) void split_p(const bf16* __restrict__ s,
                                               bf16* __restrict__ d) {
  size_t i = (size_t)blockIdx.x * blockDim.x + threadIdx.x;
  int dd = i & 63; size_t t = i >> 6; int h = t & 15; int p = (int)(t >> 4);
  d[((size_t)h * NP + p) * NDH + dd] = s[i];
}

// ---------------------------------------------------------------------------
// Fused disentangled flash attention. One wave per (b, h, 16-row q block).
// Per 16-col subtile: ctx (2 WMMA), banded c2p (4 WMMA), banded p2c (4 WMMA,
// rows indexed by k), LDS diagonal gather, online softmax; per 32 k-cols one
// PV WMMA per 16-wide d group (4 WMMA). K/pk/pq/V^T tiles stream from L2.
// ---------------------------------------------------------------------------
__global__ __launch_bounds__(32) void deberta_attn(
    const bf16* __restrict__ qg,   // (B,H,L,DH)
    const bf16* __restrict__ kg,   // (B,H,L,DH)
    const bf16* __restrict__ vtg,  // (B,H,DH,L)
    const bf16* __restrict__ pk,   // (H,P,DH)
    const bf16* __restrict__ pq,   // (H,P,DH)
    const unsigned char* __restrict__ mask,  // (B,L,L) bool
    bf16* __restrict__ out)        // (B*L, H*DH)
{
  __shared__ __attribute__((aligned(16))) float lds_ctx[16 * 16];
  __shared__ __attribute__((aligned(16))) float lds_c2p[16 * 32];
  __shared__ __attribute__((aligned(16))) float lds_p2c[16 * 32];
  __shared__ __attribute__((aligned(16))) bf16  lds_p[16 * 32];

  const float kScale = 0.07216878364870323f;  // 1/sqrt(3*64)

  int lane = threadIdx.x & 31;
  int wg = blockIdx.x;                  // b*H*64 + h*64 + qblk
  int qblk = wg & 63;
  int h = (wg >> 6) & 15;
  int b = wg >> 10;
  int q0 = qblk * 16;

  const bf16* q  = qg  + ((size_t)(b * NH + h)) * NL * NDH;
  const bf16* k  = kg  + ((size_t)(b * NH + h)) * NL * NDH;
  const bf16* vt = vtg + ((size_t)(b * NH + h)) * NDH * NL;
  const bf16* pkh = pk + (size_t)h * NP * NDH;
  const bf16* pqh = pq + (size_t)h * NP * NDH;
  const unsigned char* maskp = mask + (size_t)b * NL * NL;

  int lr = lane & 15;
  int khalf = (lane >> 4) * 8;
  int hi8 = (lane & 16) ? 8 : 0;

  // q A-fragments, loaded once (K = 64 -> two k-steps of 32)
  bf16x16 aq0 = load_frag(q + (size_t)(q0 + lr) * NDH + 0  + khalf);
  bf16x16 aq1 = load_frag(q + (size_t)(q0 + lr) * NDH + 32 + khalf);

  f32x8 o[4] = {};
  float mrow[8], lrow[8];
#pragma unroll
  for (int r = 0; r < 8; ++r) { mrow[r] = -3.0e38f; lrow[r] = 0.f; }

  for (int kt2 = 0; kt2 < NL / 32; ++kt2) {
    float pprev[8], pcur[8];
#pragma unroll
    for (int jt = 0; jt < 2; ++jt) {
      int kb = kt2 * 32 + jt * 16;
      int d0 = q0 - kb;
      int rstart = min(max(d0 + 497, 0), 1023);          // c2p band start
      int sstart = min(max(497 - d0, 0), 1023);          // p2c band start

      // --- score WMMAs ---
      bf16x16 bk0 = load_frag(k + (size_t)(kb + lr) * NDH + 0  + khalf);
      bf16x16 bk1 = load_frag(k + (size_t)(kb + lr) * NDH + 32 + khalf);
      f32x8 sctx = {};
      sctx = wmma_bf16(aq0, bk0, sctx);
      sctx = wmma_bf16(aq1, bk1, sctx);

      f32x8 c2p[2] = {};
      f32x8 p2c[2] = {};
#pragma unroll
      for (int nh = 0; nh < 2; ++nh) {
        int pr = min(rstart + nh * 16 + lr, 1023);
        bf16x16 bp0 = load_frag(pkh + (size_t)pr * NDH + 0  + khalf);
        bf16x16 bp1 = load_frag(pkh + (size_t)pr * NDH + 32 + khalf);
        c2p[nh] = wmma_bf16(aq0, bp0, c2p[nh]);
        c2p[nh] = wmma_bf16(aq1, bp1, c2p[nh]);
        int sr = min(sstart + nh * 16 + lr, 1023);
        bf16x16 bq0 = load_frag(pqh + (size_t)sr * NDH + 0  + khalf);
        bf16x16 bq1 = load_frag(pqh + (size_t)sr * NDH + 32 + khalf);
        // k_tile B-fragments double as A-fragments (identical lane layout)
        p2c[nh] = wmma_bf16(bk0, bq0, p2c[nh]);
        p2c[nh] = wmma_bf16(bk1, bq1, p2c[nh]);
      }

      // --- dump raw tiles to LDS for the diagonal-band gather ---
      __syncthreads();
#pragma unroll
      for (int r = 0; r < 8; ++r) {
        int i = r + hi8;
        lds_ctx[i * 16 + lr]      = sctx[r];
        lds_c2p[i * 32 + lr]      = c2p[0][r];
        lds_c2p[i * 32 + 16 + lr] = c2p[1][r];
        lds_p2c[i * 32 + lr]      = p2c[0][r];   // rows here are k-columns j
        lds_p2c[i * 32 + 16 + lr] = p2c[1][r];
      }
      __syncthreads();

      // --- gather + online softmax (lane owns (i = r+hi8, j = lr)) ---
      float scv[8], pm[8];
#pragma unroll
      for (int r = 0; r < 8; ++r) {
        int i = r + hi8;
        int ci = min(max(d0 + 512 + i - lr, 0), 1023) - rstart;  // in [0,31]
        int si = min(max(512 - d0 + lr - i, 0), 1023) - sstart;  // in [0,31]
        float sc = lds_ctx[i * 16 + lr] + lds_c2p[i * 32 + ci] +
                   lds_p2c[lr * 32 + si];
        sc *= kScale;
        unsigned char mk = maskp[(size_t)(q0 + i) * NL + kb + lr];
        scv[r] = mk ? sc : -10000.0f;
        pm[r] = mk ? 1.0f : 0.0f;
      }
#pragma unroll
      for (int r = 0; r < 8; ++r) {
        float t = scv[r];
#pragma unroll
        for (int off = 1; off < 16; off <<= 1)
          t = fmaxf(t, __shfl_xor(t, off, 32));       // 16-lane half reduce
        float nm = fmaxf(mrow[r], t);
        float alpha = __expf(mrow[r] - nm);
        mrow[r] = nm;
        float p = pm[r] * __expf(scv[r] - nm);
        float s = p;
#pragma unroll
        for (int off = 1; off < 16; off <<= 1)
          s += __shfl_xor(s, off, 32);
        lrow[r] = lrow[r] * alpha + s;
        pcur[r] = p;
        if (jt == 1) pprev[r] *= alpha;               // re-normalize subtile 0
#pragma unroll
        for (int nt = 0; nt < 4; ++nt) o[nt][r] *= alpha;
      }
      if (jt == 0) {
#pragma unroll
        for (int r = 0; r < 8; ++r) pprev[r] = pcur[r];
      }
    }

    // --- PV: probs(16x32) x V(32x64) ---
    __syncthreads();
#pragma unroll
    for (int r = 0; r < 8; ++r) {
      int i = r + hi8;
      lds_p[i * 32 + lr]      = (bf16)pprev[r];
      lds_p[i * 32 + 16 + lr] = (bf16)pcur[r];
    }
    __syncthreads();
    bf16x16 ap = load_frag((const bf16*)&lds_p[lr * 32 + khalf]);
#pragma unroll
    for (int nt = 0; nt < 4; ++nt) {
      bf16x16 bv = load_frag(vt + (size_t)(nt * 16 + lr) * NL +
                             kt2 * 32 + khalf);
      o[nt] = wmma_bf16(ap, bv, o[nt]);
    }
  }

  // --- finalize: divide by row sums, write (B*L, H*DH) bf16 ---
  float inv[8];
#pragma unroll
  for (int r = 0; r < 8; ++r) inv[r] = 1.0f / fmaxf(lrow[r], 1e-20f);
  int orow = q0 + hi8;
#pragma unroll
  for (int nt = 0; nt < 4; ++nt)
#pragma unroll
    for (int r = 0; r < 8; ++r)
      out[(size_t)(b * NL + orow + r) * HD + h * NDH + nt * 16 + lr] =
          (bf16)(o[nt][r] * inv[r]);
}

// ---------------------------------------------------------------------------
// x = h + proj; out = x * rsqrt(mean(x^2) + eps) * norm_w
// ---------------------------------------------------------------------------
__global__ __launch_bounds__(256) void residual_rmsnorm(
    const float* __restrict__ h, const float* __restrict__ proj,
    const float* __restrict__ w, float* __restrict__ out) {
  int row = blockIdx.x;
  const float* hr = h + (size_t)row * ND;
  const float* pr = proj + (size_t)row * ND;
  float ss = 0.f;
  for (int c = threadIdx.x; c < ND; c += 256) {
    float x = hr[c] + pr[c];
    ss += x * x;
  }
#pragma unroll
  for (int off = 16; off > 0; off >>= 1) ss += __shfl_xor(ss, off, 32);
  __shared__ float red[8];
  if ((threadIdx.x & 31) == 0) red[threadIdx.x >> 5] = ss;
  __syncthreads();
  float tot = 0.f;
#pragma unroll
  for (int i = 0; i < 8; ++i) tot += red[i];
  float inv = rsqrtf(tot * (1.0f / ND) + 1e-5f);
  float* orow = out + (size_t)row * ND;
  for (int c = threadIdx.x; c < ND; c += 256) {
    float x = hr[c] + pr[c];
    orow[c] = x * inv * w[c];
  }
}

// ---------------------------------------------------------------------------
extern "C" void kernel_launch(void* const* d_in, const int* in_sizes, int n_in,
                              void* d_out, int out_size, void* d_ws,
                              size_t ws_size, hipStream_t stream) {
  const float* h    = (const float*)d_in[0];
  const unsigned char* mask = (const unsigned char*)d_in[1];
  const float* pos  = (const float*)d_in[2];
  const float* wsrc[6] = {(const float*)d_in[3], (const float*)d_in[4],
                          (const float*)d_in[5], (const float*)d_in[6],
                          (const float*)d_in[7], (const float*)d_in[8]};
  const float* normw = (const float*)d_in[9];

  // workspace carve-out (~86 MB total)
  char* ws = (char*)d_ws;
  size_t off = 0;
  auto carve = [&](size_t bytes) -> char* {
    char* p = ws + off;
    off += (bytes + 255) & ~(size_t)255;
    return p;
  };
  const size_t M = (size_t)NB * NL;
  bf16* hB   = (bf16*)carve(M * ND * 2);
  bf16* posB = (bf16*)carve((size_t)NP * ND * 2);
  bf16* wT[6];
  for (int i = 0; i < 6; ++i) wT[i] = (bf16*)carve((size_t)ND * ND * 2);
  bf16* Ctmp = (bf16*)carve(M * HD * 2);
  bf16* qB   = (bf16*)carve(M * HD * 2);
  bf16* kB   = (bf16*)carve(M * HD * 2);
  bf16* vB   = (bf16*)carve(M * HD * 2);
  bf16* pqB  = (bf16*)carve((size_t)NH * NP * NDH * 2);
  bf16* pkB  = (bf16*)carve((size_t)NH * NP * NDH * 2);
  bf16* attnB = (bf16*)carve(M * HD * 2);
  float* proj = (float*)carve(M * ND * 4);

  // 1) converts
  cvt_f32_bf16<<<512, 256, 0, stream>>>(h, hB, (int)(M * ND));
  cvt_f32_bf16<<<256, 256, 0, stream>>>(pos, posB, NP * ND);
  // 2) weight transposes (NT GEMM operand layout)
  dim3 tb(32, 8), tg(ND / 32, ND / 32);
  for (int i = 0; i < 6; ++i)
    transpose_cvt<<<tg, tb, 0, stream>>>(wsrc[i], wT[i]);

  dim3 gblk(256);
  dim3 gh((unsigned)(M / 128), ND / 64);   // (32,16) for h-projections
  dim3 gp(NP / 128, ND / 64);              // (8,16)  for position projections
  int nsplit = (int)(M * HD / 256);

  // 3) projections + head splits
  gemm_nt_wmma<false><<<gh, gblk, 0, stream>>>(hB, wT[0], Ctmp, (int)M, HD, ND);
  split_qk<<<nsplit, 256, 0, stream>>>(Ctmp, qB);
  gemm_nt_wmma<false><<<gh, gblk, 0, stream>>>(hB, wT[1], Ctmp, (int)M, HD, ND);
  split_qk<<<nsplit, 256, 0, stream>>>(Ctmp, kB);
  gemm_nt_wmma<false><<<gh, gblk, 0, stream>>>(hB, wT[2], Ctmp, (int)M, HD, ND);
  split_v<<<nsplit, 256, 0, stream>>>(Ctmp, vB);
  gemm_nt_wmma<false><<<gp, gblk, 0, stream>>>(posB, wT[3], Ctmp, NP, HD, ND);
  split_p<<<NP * HD / 256, 256, 0, stream>>>(Ctmp, pqB);
  gemm_nt_wmma<false><<<gp, gblk, 0, stream>>>(posB, wT[4], Ctmp, NP, HD, ND);
  split_p<<<NP * HD / 256, 256, 0, stream>>>(Ctmp, pkB);

  // 4) fused disentangled flash attention: one wave per (b,h,16 q rows)
  deberta_attn<<<NB * NH * (NL / 16), 32, 0, stream>>>(qB, kB, vB, pkB, pqB,
                                                       mask, attnB);

  // 5) output projection + residual RMS norm
  gemm_nt_wmma<true><<<gh, gblk, 0, stream>>>(attnB, wT[5], proj, (int)M, ND, HD);
  residual_rmsnorm<<<(unsigned)M, 256, 0, stream>>>(h, proj, normw,
                                                    (float*)d_out);
}
